// HRMChess_74775380623564
// MI455X (gfx1250) — matrix operand order repaired and checked by reading, not compiled, not run on hardware
//
#include <hip/hip_runtime.h>
#include <hip/hip_bf16.h>

typedef unsigned short u16;
typedef __attribute__((ext_vector_type(16))) __bf16 v16bf;
typedef __attribute__((ext_vector_type(8)))  float  v8f;

// ---- CDNA5 async-to-LDS staging (ASYNCcnt path) via inline asm ----
// The clang builtin exists but is declared with CUDA-lang address spaces that
// generic HIP pointers don't convert to; inline asm sidesteps Sema entirely.
// Set USE_ASYNC to 0 to fall back to synchronous VGPR staging.
#define USE_ASYNC 1

#if USE_ASYNC
__device__ __forceinline__ void cp16(u16* l, const u16* g) {
  unsigned lds = (unsigned)(size_t)l;   // AS3 offset lives in low 32 bits of
                                        // a generic pointer into LDS
  asm volatile("global_load_async_to_lds_b128 %0, %1, off"
               :: "v"(lds), "v"(g) : "memory");
}
#define WAIT_ASYNC(n) asm volatile("s_wait_asynccnt " #n ::: "memory")
#else
__device__ __forceinline__ void cp16(u16* l, const u16* g) {
  *(uint4*)l = *(const uint4*)g;
}
#define WAIT_ASYNC(n) ((void)0)
#endif

__device__ __forceinline__ u16 f2bf(float f) {
  unsigned u = __float_as_uint(f);
  u += 0x7FFF + ((u >> 16) & 1);           // round to nearest even
  return (u16)(u >> 16);
}

// -------------------- prep kernels --------------------

// bitplanes [B,20,8,8] f32 -> A0 [B*64, 32] bf16 (channel-inner, padded 20->32)
__global__ void k_bitplanes(const float* __restrict__ src, u16* __restrict__ dst) {
  int tid = blockIdx.x * blockDim.x + threadIdx.x;   // P*32 threads, exact
  int r = tid >> 5, c = tid & 31;
  int b = r >> 6, pos = r & 63;
  float v = (c < 20) ? src[(size_t)(b * 20 + c) * 64 + pos] : 0.f;
  dst[(size_t)r * 32 + c] = f2bf(v);
}

// conv weight OIHW f32 -> [9][Kpad][Cout] bf16 (tap-major, K padded)
__global__ void k_convw(const float* __restrict__ src, u16* __restrict__ dst,
                        int Cin, int Kpad, int Cout) {
  int tid = blockIdx.x * blockDim.x + threadIdx.x;
  if (tid >= 9 * Kpad * Cout) return;
  int t = tid / (Kpad * Cout);
  int rem = tid - t * (Kpad * Cout);
  int k = rem / Cout;
  int n = rem - k * Cout;
  float v = (k < Cin) ? src[(size_t)(n * Cin + k) * 9 + t] : 0.f;
  dst[tid] = f2bf(v);                       // dst[(t*Kpad+k)*Cout+n] == dst[tid]
}

__global__ void k_cvt(const float* __restrict__ src, u16* __restrict__ dst, int n) {
  int tid = blockIdx.x * blockDim.x + threadIdx.x;
  if (tid < n) dst[tid] = f2bf(src[tid]);
}

__global__ void k_zero(u16* __restrict__ dst, int n) {
  int tid = blockIdx.x * blockDim.x + threadIdx.x;
  if (tid < n) dst[tid] = 0;
}

// LayerNorm over 256 features: one wave32 per row, f32 in -> bf16 out
__global__ void k_ln(const float* __restrict__ x, const float* __restrict__ g,
                     const float* __restrict__ b, u16* __restrict__ out) {
  int lane = threadIdx.x & 31;
  int row  = blockIdx.x * 8 + (threadIdx.x >> 5);
  const float* xr = x + (size_t)row * 256;
  float v[8], s = 0.f, sq = 0.f;
#pragma unroll
  for (int i = 0; i < 8; ++i) { v[i] = xr[lane + 32 * i]; s += v[i]; sq += v[i] * v[i]; }
#pragma unroll
  for (int m = 16; m >= 1; m >>= 1) { s += __shfl_xor(s, m, 32); sq += __shfl_xor(sq, m, 32); }
  float mean = s * (1.f / 256.f);
  float rstd = rsqrtf(sq * (1.f / 256.f) - mean * mean + 1e-5f);
#pragma unroll
  for (int i = 0; i < 8; ++i) {
    int c = lane + 32 * i;
    out[(size_t)row * 256 + c] = f2bf((v[i] - mean) * rstd * g[c] + b[c]);
  }
}

// -------------------- WMMA GEMM (double-buffered async pipeline) --------------------
// C[M,N] = act(A[M,K] @ B[K,N] + bias)
// CONV=1: A is [M, K] position-major activation, Bw is [9][K][N]; 9-tap shifted
//         accumulation (SAME 3x3 conv over 8x8 boards, rows = b*64 + y*8 + x).
//         Halo: A addresses clamped in-bounds at load; out-of-bounds rows are
//         zeroed in the A *fragments* (each lane owns one matrix row).
// Segmented A (CONV=0): k -> segment k/segLen selects A0/A1/A2 (concat inputs).
// ACT: 0=none 1=relu 2=gelu(exact).  OUT: 0=bf16 row-major, 1=f32 row-major,
// 2=bf16 NCHW-flatten scatter (feature = n*64 + pos) for the conv3 -> MLP seam.

union FragAB { v16bf v; unsigned u[8]; };

template <int CONV, int ACT, int OUT>
__global__ __launch_bounds__(256) void gemm_wmma(
    const u16* __restrict__ A0, const u16* __restrict__ A1, const u16* __restrict__ A2,
    int segLen, const u16* __restrict__ Bw, const float* __restrict__ bias,
    void* __restrict__ Cout, int M, int N, int K) {
  __shared__ __align__(16) u16 lsA[2][128 * 40];  // 128 rows x 32 k, stride 40 (pad)
  __shared__ __align__(16) u16 lsB[2][32 * 72];   // 32 k x 64 n, stride 72 (pad)

  const int tid  = threadIdx.x;
  const int lane = tid & 31, wv = tid >> 5;
  const int wm   = (wv >> 1) * 32;                // wave M offset in block tile
  const int wn   = (wv & 1) * 32;                 // wave N offset
  const int half = lane >> 4, mrow = lane & 15;
  const int m0   = blockIdx.x * 128;
  const int nb0  = blockIdx.y * 64;

  const int Kst = K >> 5;                         // 32-wide k-steps per tap
  const int S   = (CONV ? 9 : 1) * Kst;           // total pipeline stages
  const size_t KN = (size_t)K * (size_t)N;

  // per-thread staging coordinates
  const int lr   = tid >> 1;                      // A tile row this thread stages
  const int colh = (tid & 1) * 16;                // which 16-elem half of the row
  const int eB   = tid * 8, kr = eB >> 6, nc = eB & 63;

  v8f acc[2][2];
  const v8f vz = {0.f, 0.f, 0.f, 0.f, 0.f, 0.f, 0.f, 0.f};
#pragma unroll
  for (int i = 0; i < 2; ++i)
#pragma unroll
    for (int j = 0; j < 2; ++j) acc[i][j] = vz;

  // issue one stage's tile loads into ping-pong buffer pb (uniform 3 async ops)
  auto issue = [&](int s, int pb) {
    const int tap = CONV ? (s / Kst) : 0;
    const int k0  = (s - tap * Kst) * 32;
    // ---- A half-row ----
    int kk = k0 + colh;
    const u16* seg = A0; int ko = kk;
    if (!CONV && kk >= segLen) {
      if (kk >= 2 * segLen) { seg = A2; ko = kk - 2 * segLen; }
      else                  { seg = A1; ko = kk - segLen; }
    }
    const int r = m0 + lr;
    size_t arow;
    if (CONV) {
      const int ty = tap / 3 - 1, tx = tap % 3 - 1;
      int b = r >> 6, pos = r & 63;
      int yy = (pos >> 3) + ty, xx = (pos & 7) + tx;
      yy = yy < 0 ? 0 : (yy > 7 ? 7 : yy);        // clamp; halo zeroed in frags
      xx = xx < 0 ? 0 : (xx > 7 ? 7 : xx);
      arow = (size_t)(b * 64 + yy * 8 + xx);
    } else {
      arow = (size_t)r;
    }
    const u16* ga = seg + arow * (size_t)segLen + ko;
    u16* la = &lsA[pb][lr * 40 + colh];
    cp16(la, ga);
    cp16(la + 8, ga + 8);
    // ---- B ----
    const u16* gb = (CONV ? Bw + (size_t)tap * KN : Bw) +
                    (size_t)(k0 + kr) * N + nb0 + nc;
    cp16(&lsB[pb][kr * 72 + nc], gb);
  };

  issue(0, 0);
  for (int s = 0; s < S; ++s) {
    const int pb = s & 1;
    if (s + 1 < S) { issue(s + 1, pb ^ 1); WAIT_ASYNC(3); }  // drain stage s only
    else           { WAIT_ASYNC(0); }
    __syncthreads();

    const int tap = CONV ? (s / Kst) : 0;
    FragAB fa[2], fb[2];
#pragma unroll
    for (int mi = 0; mi < 2; ++mi) {
      const int rowm  = wm + mi * 16 + mrow;
      const int rbase = rowm * 40;
#pragma unroll
      for (int j = 0; j < 8; ++j) {
        // A 16x32 bf16: VGPR j<4 -> K = half*8 + 2j; j>=4 -> 16 + half*8 + 2(j-4)
        int k = (j < 4) ? (half * 8 + 2 * j) : (16 + half * 8 + 2 * (j - 4));
        fa[mi].u[j] = *(const unsigned*)&lsA[pb][rbase + k];
      }
      if (CONV) {                                  // SAME padding: zero halo rows
        int pos = (m0 + rowm) & 63;
        int yy = (pos >> 3) + tap / 3 - 1, xx = (pos & 7) + tap % 3 - 1;
        if (!(((unsigned)yy < 8u) && ((unsigned)xx < 8u))) {
#pragma unroll
          for (int j = 0; j < 8; ++j) fa[mi].u[j] = 0;
        }
      }
    }
#pragma unroll
    for (int ni = 0; ni < 2; ++ni) {
      const int cbase = wn + ni * 16 + mrow;  // B 32x16: N = lane&15, K as A
#pragma unroll
      for (int v = 0; v < 8; ++v) {
        int k = (v < 4) ? (half * 8 + 2 * v) : (16 + half * 8 + 2 * (v - 4));
        fb[ni].u[v] = (unsigned)lsB[pb][cbase + k * 72] |
                      ((unsigned)lsB[pb][cbase + (k + 1) * 72] << 16);
      }
    }

#pragma unroll
    for (int mi = 0; mi < 2; ++mi)
#pragma unroll
      for (int ni = 0; ni < 2; ++ni)
        acc[mi][ni] = __builtin_amdgcn_wmma_f32_16x16x32_bf16(
            false, fa[mi].v, false, fb[ni].v, (short)0, acc[mi][ni], false, false);
    __syncthreads();                               // protect ping-pong reuse
  }

  // ---- epilogue: C layout VGPR r -> M = r + half*8, N = lane&15 ----
#pragma unroll
  for (int mi = 0; mi < 2; ++mi) {
#pragma unroll
    for (int ni = 0; ni < 2; ++ni) {
      int gn = nb0 + wn + ni * 16 + mrow;
      float bv = bias[gn];
#pragma unroll
      for (int r = 0; r < 8; ++r) {
        int gm = m0 + wm + mi * 16 + half * 8 + r;
        float vv = acc[mi][ni][r] + bv;
        if (ACT == 1)      vv = fmaxf(vv, 0.f);
        else if (ACT == 2) vv = 0.5f * vv * (1.f + erff(vv * 0.70710678118f));
        if (OUT == 0)      ((u16*)Cout)[(size_t)gm * N + gn] = f2bf(vv);
        else if (OUT == 1) ((float*)Cout)[(size_t)gm * N + gn] = vv;
        else {
          int b = gm >> 6, pos = gm & 63;          // NCHW flatten: feature = n*64+pos
          ((u16*)Cout)[(size_t)b * 4096 + gn * 64 + pos] = f2bf(vv);
        }
      }
    }
  }
}

// -------------------- host --------------------

extern "C" void kernel_launch(void* const* d_in, const int* in_sizes, int n_in,
                              void* d_out, int out_size, void* d_ws, size_t ws_size,
                              hipStream_t stream) {
  (void)in_sizes; (void)n_in; (void)out_size; (void)ws_size;
  const int Bn = 4096;
  const size_t P = (size_t)Bn * 64;            // 262144 board positions

  const float* bitplanes = (const float*)d_in[0];
  const float* conv_w1 = (const float*)d_in[1];  const float* conv_b1 = (const float*)d_in[2];
  const float* conv_w2 = (const float*)d_in[3];  const float* conv_b2 = (const float*)d_in[4];
  const float* conv_w3 = (const float*)d_in[5];  const float* conv_b3 = (const float*)d_in[6];
  const float* enh_b1 = (const float*)d_in[8];
  const float* enh_b2 = (const float*)d_in[10];
  const float* l_b1 = (const float*)d_in[12];
  const float* l_b2 = (const float*)d_in[14];
  const float* l_g  = (const float*)d_in[15];    const float* l_bt = (const float*)d_in[16];
  const float* h_b1 = (const float*)d_in[18];
  const float* h_b2 = (const float*)d_in[20];
  const float* h_g  = (const float*)d_in[21];    const float* h_bt = (const float*)d_in[22];
  const float* v_b1 = (const float*)d_in[24];
  const float* v_b2 = (const float*)d_in[26];

  char* ws = (char*)d_ws;
  size_t off = 0;
  auto alloc = [&](size_t bytes) -> char* {
    char* p = ws + off; off = (off + bytes + 255) & ~(size_t)255; return p;
  };
  u16* A0    = (u16*)alloc(P * 32 * 2);
  u16* act1  = (u16*)alloc(P * 256 * 2);
  u16* act2  = (u16*)alloc(P * 128 * 2);
  u16* act3  = (u16*)alloc((size_t)Bn * 4096 * 2);
  u16* be    = (u16*)alloc((size_t)Bn * 256 * 2);
  u16* zl    = (u16*)alloc((size_t)Bn * 256 * 2);
  u16* zh    = (u16*)alloc((size_t)Bn * 256 * 2);
  u16* t1    = (u16*)alloc((size_t)Bn * 256 * 2);
  float* t2f = (float*)alloc((size_t)Bn * 256 * 4);
  u16* wc1   = (u16*)alloc((size_t)9 * 32 * 256 * 2);
  u16* wc2   = (u16*)alloc((size_t)9 * 256 * 128 * 2);
  u16* wc3   = (u16*)alloc((size_t)9 * 128 * 64 * 2);
  u16* wenh1 = (u16*)alloc((size_t)4096 * 256 * 2);
  u16* wenh2 = (u16*)alloc((size_t)256 * 256 * 2);
  u16* wl1   = (u16*)alloc((size_t)768 * 256 * 2);
  u16* wl2   = (u16*)alloc((size_t)256 * 256 * 2);
  u16* wh1   = (u16*)alloc((size_t)512 * 256 * 2);
  u16* wh2   = (u16*)alloc((size_t)256 * 256 * 2);
  u16* wv1   = (u16*)alloc((size_t)256 * 256 * 2);
  u16* wv2   = (u16*)alloc((size_t)256 * 128 * 2);

  // ---- prep: bf16 conversions / layout reorgs ----
  k_bitplanes<<<(unsigned)(P * 32 / 256), 256, 0, stream>>>(bitplanes, A0);
  k_convw<<<(9 * 32 * 256 + 255) / 256, 256, 0, stream>>>(conv_w1, wc1, 20, 32, 256);
  k_convw<<<(9 * 256 * 128 + 255) / 256, 256, 0, stream>>>(conv_w2, wc2, 256, 256, 128);
  k_convw<<<(9 * 128 * 64 + 255) / 256, 256, 0, stream>>>(conv_w3, wc3, 128, 128, 64);
  auto cvt = [&](const void* s, u16* d, int n) {
    k_cvt<<<(n + 255) / 256, 256, 0, stream>>>((const float*)s, d, n);
  };
  cvt(d_in[7],  wenh1, 4096 * 256);
  cvt(d_in[9],  wenh2, 256 * 256);
  cvt(d_in[11], wl1,   768 * 256);
  cvt(d_in[13], wl2,   256 * 256);
  cvt(d_in[17], wh1,   512 * 256);
  cvt(d_in[19], wh2,   256 * 256);
  cvt(d_in[23], wv1,   256 * 256);
  cvt(d_in[25], wv2,   256 * 128);

  const unsigned MBp = (unsigned)(P / 128);    // 2048 M-tiles for conv GEMMs

  // ---- conv tower (shifted-tap WMMA GEMMs, relu) ----
  gemm_wmma<1, 1, 0><<<dim3(MBp, 4), 256, 0, stream>>>(A0, nullptr, nullptr, 32,  wc1, conv_b1, act1, (int)P, 256, 32);
  gemm_wmma<1, 1, 0><<<dim3(MBp, 2), 256, 0, stream>>>(act1, nullptr, nullptr, 256, wc2, conv_b2, act2, (int)P, 128, 256);
  gemm_wmma<1, 1, 2><<<dim3(MBp, 1), 256, 0, stream>>>(act2, nullptr, nullptr, 128, wc3, conv_b3, act3, (int)P, 64, 128);

  // ---- board encoder MLP ----
  gemm_wmma<0, 1, 0><<<dim3(32, 4), 256, 0, stream>>>(act3, nullptr, nullptr, 4096, wenh1, enh_b1, t1, Bn, 256, 4096);
  gemm_wmma<0, 1, 0><<<dim3(32, 4), 256, 0, stream>>>(t1, nullptr, nullptr, 256, wenh2, enh_b2, be, Bn, 256, 256);

  k_zero<<<(Bn * 256 + 255) / 256, 256, 0, stream>>>(zl, Bn * 256);
  k_zero<<<(Bn * 256 + 255) / 256, 256, 0, stream>>>(zh, Bn * 256);

  // ---- HRM recurrence: 8 cycles x (8 l-steps + 1 h-step) ----
  for (int c = 0; c < 8; ++c) {
    for (int t = 0; t < 8; ++t) {
      gemm_wmma<0, 1, 0><<<dim3(32, 4), 256, 0, stream>>>(zl, zh, be, 256, wl1, l_b1, t1, Bn, 256, 768);
      gemm_wmma<0, 1, 1><<<dim3(32, 4), 256, 0, stream>>>(t1, nullptr, nullptr, 256, wl2, l_b2, t2f, Bn, 256, 256);
      k_ln<<<Bn / 8, 256, 0, stream>>>(t2f, l_g, l_bt, zl);
    }
    gemm_wmma<0, 1, 0><<<dim3(32, 4), 256, 0, stream>>>(zh, zl, nullptr, 256, wh1, h_b1, t1, Bn, 256, 512);
    gemm_wmma<0, 1, 1><<<dim3(32, 4), 256, 0, stream>>>(t1, nullptr, nullptr, 256, wh2, h_b2, t2f, Bn, 256, 256);
    k_ln<<<Bn / 8, 256, 0, stream>>>(t2f, h_g, h_bt, zh);
  }

  // ---- value head: gelu MLP -> f32 logits [4096,128] ----
  gemm_wmma<0, 2, 0><<<dim3(32, 4), 256, 0, stream>>>(zh, nullptr, nullptr, 256, wv1, v_b1, t1, Bn, 256, 256);
  gemm_wmma<0, 0, 1><<<dim3(32, 2), 256, 0, stream>>>(t1, nullptr, nullptr, 256, wv2, v_b2, d_out, Bn, 128, 256);
}
